// LSTM_34084860461176
// MI455X (gfx1250) — matrix operand order repaired and checked by reading, not compiled, not run on hardware
//
#include <hip/hip_runtime.h>
#include <stdint.h>

// Problem constants (match reference)
#define TT 512
#define BB 64
#define II 1024
#define HH 1024
#define TBI (TT*BB)   // 32768 rows of the big input GEMM

typedef __attribute__((ext_vector_type(16))) __bf16 v16bf;
typedef __attribute__((ext_vector_type(8)))  float  v8f;
typedef __attribute__((ext_vector_type(4)))  uint32_t v4u;
typedef __attribute__((ext_vector_type(8)))  int      v8i;
typedef __attribute__((ext_vector_type(4)))  int      v4i;

union AB16 { v16bf v; uint32_t u[8]; uint4 q[2]; };

__device__ __forceinline__ uint16_t f2bf(float f){
  uint32_t u = __builtin_bit_cast(uint32_t, f);
  u += 0x7FFFu + ((u >> 16) & 1u);          // round-to-nearest-even
  return (uint16_t)(u >> 16);
}
__device__ __forceinline__ uint32_t pack2(float a, float b){
  return (uint32_t)f2bf(a) | ((uint32_t)f2bf(b) << 16);
}
// A-fragment K offset within a 32-wide K chunk (ISA 7.12.2, 16-bit A 16x32)
__device__ __forceinline__ int a_koff(int v, int khalf){
  return (v < 4 ? 0 : 16) + khalf * 8 + (v & 3) * 2;
}

#if __has_builtin(__builtin_amdgcn_tensor_load_to_lds) && __has_builtin(__builtin_amdgcn_s_wait_tensorcnt)
#define USE_TDM 1
#endif

#ifdef USE_TDM
// Issue one TDM 2-D tile load: tile_d0 x tile_d1 elements of 2 bytes,
// from a row-major tensor with row stride stride_d0 (elements), into
// contiguous LDS (rows of tile_d0*2 bytes). ISA 8.3/8.4 D# layout.
// This toolchain exposes the 6-arg builtin:
//   (uint32x4 g0, int32x8 g1, int32x4 g2, int32x4 g3, int32x8 gx, i32 cpol)
// 2-D tensors: groups 2/3 (and the trailing group) are zero/NULL.
__device__ __forceinline__ void tdm_load_2d(const void* gsrc, void* ldst,
                                            uint32_t tensor_d0, uint32_t tensor_d1,
                                            uint32_t stride_d0,
                                            uint32_t tile_d0, uint32_t tile_d1){
  uint64_t ga = (uint64_t)(uintptr_t)gsrc;
  uint32_t la = (uint32_t)(uintptr_t)ldst;   // wave32 generic->LDS: low 32 bits are the LDS byte offset
  v4u g0;
  g0.x = 1u;                                           // count=1, user mode
  g0.y = la;                                           // lds_addr
  g0.z = (uint32_t)ga;                                 // global_addr[31:0]
  g0.w = (uint32_t)((ga >> 32) & 0x01FFFFFFu) | (2u << 30);  // addr[56:32] | type=2
  v8i g1;
  g1[0] = (int)(1u << 16);                             // data_size = 1 -> 2 bytes
  g1[1] = (int)((tensor_d0 & 0xFFFFu) << 16);          // tensor_dim0[15:0]
  g1[2] = (int)((tensor_d0 >> 16) | ((tensor_d1 & 0xFFFFu) << 16)); // dim0 hi | dim1 lo
  g1[3] = (int)((tensor_d1 >> 16) | (tile_d0 << 16));  // dim1 hi | tile_dim0
  g1[4] = (int)tile_d1;                                // tile_dim1, tile_dim2=0
  g1[5] = (int)stride_d0;                              // tensor_dim0_stride[31:0]
  g1[6] = 0;                                           // stride hi, dim1_stride lo
  g1[7] = 0;
  v4i z4 = {0, 0, 0, 0};
  v8i z8 = {0, 0, 0, 0, 0, 0, 0, 0};
  __builtin_amdgcn_tensor_load_to_lds(g0, g1, z4, z4, z8, 0);
}
#endif

// ---------------- prep kernels ----------------

// xg[g][r][i] = bf16( x[r][i] * maskx[g*B + (r%B)][i] ), pairs of 2 elems/thread
__global__ __launch_bounds__(256) void k_prep_x(const float* __restrict__ x,
                                                const float* __restrict__ maskx,
                                                uint16_t* __restrict__ xg){
  size_t e = ((size_t)blockIdx.x * blockDim.x + threadIdx.x) * 2;
  const size_t per_g = (size_t)TBI * II;
  if (e >= 4 * per_g) return;
  int g = (int)(e / per_g);
  size_t rem = e - (size_t)g * per_g;
  size_t r = rem / II;
  int i = (int)(rem - r * II);
  int b = (int)(r % BB);
  const float* mrow = maskx + ((size_t)(g * BB + b)) * II;
  const float* xrow = x + r * II;
  *(uint32_t*)(xg + e) = pack2(xrow[i] * mrow[i], xrow[i + 1] * mrow[i + 1]);
}

// wT[g][n][k] = bf16( w[g][k][n] )   (K = N = 1024)
__global__ __launch_bounds__(256) void k_prep_wT(const float* __restrict__ w,
                                                 uint16_t* __restrict__ wT){
  size_t idx = (size_t)blockIdx.x * blockDim.x + threadIdx.x;
  if (idx >= (size_t)4 * 1024 * 1024) return;
  int g = (int)(idx >> 20);
  int n = (int)((idx >> 10) & 1023);
  int k = (int)(idx & 1023);
  wT[idx] = f2bf(w[(((size_t)g << 10) + k) * 1024 + n]);
}

__global__ __launch_bounds__(256) void k_prep_bias(const float* __restrict__ bx,
                                                   const float* __restrict__ bh,
                                                   float* __restrict__ bias){
  int idx = blockIdx.x * blockDim.x + threadIdx.x;
  if (idx < 4 * HH) bias[idx] = bx[idx] + bh[idx];
}

// hm[g][b][n] = bf16(h0*maskh); cstate = c0
__global__ __launch_bounds__(256) void k_init_state(const float* __restrict__ h0,
                                                    const float* __restrict__ c0,
                                                    const float* __restrict__ maskh,
                                                    uint16_t* __restrict__ hm,
                                                    float* __restrict__ cstate){
  int idx = blockIdx.x * blockDim.x + threadIdx.x;     // over 4*B*H
  if (idx >= 4 * BB * HH) return;
  int g = idx / (BB * HH);
  int bn = idx - g * (BB * HH);
  hm[idx] = f2bf(h0[bn] * maskh[idx]);
  if (g == 0) cstate[bn] = c0[bn];
}

// ---------------- phase 1: big input GEMM ----------------
// gxall[t][g][b][n] = sum_k xg[g][t*B+b][k] * wxT[g][n][k] + bias[g][n]
// Block = 256 threads (8 waves). Block tile M=128 (wave strips of 16), N=64, K chunk 32.
// A and B tiles staged in LDS by the Tensor Data Mover, double buffered.
#define BK 32
__global__ __launch_bounds__(256) void k_gemm_x(const uint16_t* __restrict__ xg,
                                                const uint16_t* __restrict__ wxT,
                                                const float* __restrict__ bias,
                                                float* __restrict__ gxall){
  __shared__ __align__(16) uint16_t sA[2][128 * BK];   // [m][k] rows of 64B
  __shared__ __align__(16) uint16_t sB[2][64 * BK];    // [n][k] rows of 64B

  const int tid  = threadIdx.x;
  const int wave = tid >> 5;
  const int lane = tid & 31;
  const int l15  = lane & 15;
  const int khalf = lane >> 4;
  const int g  = blockIdx.z;
  const int mblk = blockIdx.x * 128;
  const int n0 = blockIdx.y * 64;

  const uint16_t* Ag = xg  + (size_t)g * TBI * II;     // [TBI][II]
  const uint16_t* Bg = wxT + (size_t)g * HH * II;      // [HH][II]

  v8f acc[4];
  #pragma unroll
  for (int nt = 0; nt < 4; ++nt){
    float bv = bias[g * HH + n0 + nt * 16 + l15];      // l15 = output column
    #pragma unroll
    for (int v = 0; v < 8; ++v) acc[nt][v] = bv;
  }

  const int nch = II / BK;                              // 32 K-chunks

#ifdef USE_TDM
  // Prologue: wave 0 issues TDM loads for chunk 0 into buffer 0.
  if (tid < 32){
    tdm_load_2d(Ag + (size_t)mblk * II, &sA[0][0], II, TBI, II, BK, 128);
    tdm_load_2d(Bg + (size_t)n0   * II, &sB[0][0], II, HH,  II, BK, 64);
  }
#else
  { // synchronous cooperative stage of chunk 0
    int ro = tid >> 1, part = tid & 1;                  // A: 128 rows x 2 halves of 16B
    *(uint4*)&sA[0][ro * BK + part * 8] =
        *(const uint4*)(Ag + (size_t)(mblk + ro) * II + part * 8);
    if (tid < 128){
      int r2 = tid >> 1, p2 = tid & 1;                  // B: 64 rows x 2 halves
      *(uint4*)&sB[0][r2 * BK + p2 * 8] =
          *(const uint4*)(Bg + (size_t)(n0 + r2) * II + p2 * 8);
    }
  }
#endif

  for (int i = 0; i < nch; ++i){
    const int buf = i & 1;
    const int k0n = (i + 1) * BK;
#ifdef USE_TDM
    if (tid < 32){
      if (i + 1 < nch){
        tdm_load_2d(Ag + (size_t)mblk * II + k0n, &sA[buf ^ 1][0], II, TBI, II, BK, 128);
        tdm_load_2d(Bg + (size_t)n0   * II + k0n, &sB[buf ^ 1][0], II, HH,  II, BK, 64);
        __builtin_amdgcn_s_wait_tensorcnt(2);   // chunk i landed (in-order completion)
      } else {
        __builtin_amdgcn_s_wait_tensorcnt(0);
      }
    }
#else
    if (i + 1 < nch){
      int ro = tid >> 1, part = tid & 1;
      *(uint4*)&sA[buf ^ 1][ro * BK + part * 8] =
          *(const uint4*)(Ag + (size_t)(mblk + ro) * II + k0n + part * 8);
      if (tid < 128){
        int r2 = tid >> 1, p2 = tid & 1;
        *(uint4*)&sB[buf ^ 1][r2 * BK + p2 * 8] =
            *(const uint4*)(Bg + (size_t)(n0 + r2) * II + k0n + p2 * 8);
      }
    }
#endif
    __syncthreads();   // chunk i visible to all waves

    // Hoist ALL fragment loads (10x ds_load_b128) ahead of the WMMA burst so
    // the 4 WMMAs issue back-to-back (independent accumulators, shared A).
    AB16 a, b[4];
    const uint16_t* ar = &sA[buf][(wave * 16 + l15) * BK];
    a.q[0] = *(const uint4*)(ar + khalf * 8);
    a.q[1] = *(const uint4*)(ar + 16 + khalf * 8);
    #pragma unroll
    for (int nt = 0; nt < 4; ++nt){
      const uint16_t* br = &sB[buf][(nt * 16 + l15) * BK];
      b[nt].q[0] = *(const uint4*)(br + khalf * 8);
      b[nt].q[1] = *(const uint4*)(br + 16 + khalf * 8);
    }
    #pragma unroll
    for (int nt = 0; nt < 4; ++nt)
      acc[nt] = __builtin_amdgcn_wmma_f32_16x16x32_bf16(
          false, a.v, false, b[nt].v, (short)0, acc[nt], false, false);

    __syncthreads();   // protect buffer from next prefetch overwrite
  }

  const int m0 = mblk + wave * 16;
  #pragma unroll
  for (int nt = 0; nt < 4; ++nt){
    int col = n0 + nt * 16 + l15;
    #pragma unroll
    for (int v = 0; v < 8; ++v){
      int m = m0 + v + 8 * khalf;
      int t = m / BB, b = m % BB;
      gxall[(((size_t)t * 4 + g) * BB + b) * HH + col] = acc[nt][v];
    }
  }
}

// ---------------- phase 2a: recurrent GEMM (one timestep) ----------------
// gxt[g][b][n] += sum_k hm[g][b][k] * whT[g][n][k]   (in place, fp32 accum)
// grid (1, H/16, 4); 4 waves cover M = 64 rows. Latency-bound: direct global loads,
// whT (8MB) and hm (512KB) are L2-resident.
__global__ __launch_bounds__(128) void k_gemm_h(const uint16_t* __restrict__ hm,
                                                const uint16_t* __restrict__ whT,
                                                float* __restrict__ gxt){
  const int wave = threadIdx.x >> 5;
  const int lane = threadIdx.x & 31;
  const int l15  = lane & 15;
  const int khalf = lane >> 4;
  const int g  = blockIdx.z;
  const int m0 = wave * 16;
  const int n0 = blockIdx.y * 16;

  const uint32_t* Arow = (const uint32_t*)(hm + (size_t)(g * BB + m0 + l15) * HH);
  const uint32_t* Brow = (const uint32_t*)(whT + ((size_t)g * HH + n0 + l15) * HH);

  v8f acc;
  #pragma unroll
  for (int v = 0; v < 8; ++v)
    acc[v] = gxt[((size_t)g * BB + m0 + v + 8 * khalf) * HH + n0 + l15];

  for (int k0 = 0; k0 < HH; k0 += 32){
    AB16 a, b;
    #pragma unroll
    for (int v = 0; v < 8; ++v) a.u[v] = Arow[(k0 + a_koff(v, khalf)) >> 1];
    #pragma unroll
    for (int v = 0; v < 8; ++v) b.u[v] = Brow[(k0 + a_koff(v, khalf)) >> 1];
    acc = __builtin_amdgcn_wmma_f32_16x16x32_bf16(
        false, a.v, false, b.v, (short)0, acc, false, false);
  }

  #pragma unroll
  for (int v = 0; v < 8; ++v)
    gxt[((size_t)g * BB + m0 + v + 8 * khalf) * HH + n0 + l15] = acc[v];
}

// ---------------- phase 2b: activations + next-step operand prep ----------------
__global__ __launch_bounds__(256) void k_act(const float* __restrict__ gxt,
                                             const float* __restrict__ maskh,
                                             float* __restrict__ cstate,
                                             uint16_t* __restrict__ hm,
                                             float* __restrict__ out_t,
                                             float* __restrict__ hf,
                                             float* __restrict__ cf,
                                             int last){
  int idx = blockIdx.x * blockDim.x + threadIdx.x;   // over B*H
  if (idx >= BB * HH) return;
  const int BH = BB * HH;
  float gi = gxt[0 * BH + idx];
  float gf = gxt[1 * BH + idx];
  float go = gxt[2 * BH + idx];
  float gn = gxt[3 * BH + idx];
  float i_g = 1.f / (1.f + expf(-gi));
  float f_g = 1.f / (1.f + expf(-gf));
  float o_g = 1.f / (1.f + expf(-go));
  float n_g = tanhf(gn);
  float c = f_g * cstate[idx] + i_g * n_g;
  float h = o_g * tanhf(c);
  cstate[idx] = c;
  out_t[idx]  = h;
  #pragma unroll
  for (int g = 0; g < 4; ++g)
    hm[g * BH + idx] = f2bf(h * maskh[g * BH + idx]);
  if (last){ hf[idx] = h; cf[idx] = c; }
}

// ---------------- host launch ----------------
extern "C" void kernel_launch(void* const* d_in, const int* in_sizes, int n_in,
                              void* d_out, int out_size, void* d_ws, size_t ws_size,
                              hipStream_t stream) {
  const float* x     = (const float*)d_in[0];
  const float* h0    = (const float*)d_in[1];
  const float* c0    = (const float*)d_in[2];
  const float* maskx = (const float*)d_in[3];
  const float* maskh = (const float*)d_in[4];
  const float* wx    = (const float*)d_in[5];
  const float* wh    = (const float*)d_in[6];
  const float* bx    = (const float*)d_in[7];
  const float* bh    = (const float*)d_in[8];
  float* out = (float*)d_out;

  // workspace carve (256B aligned)
  char* ws = (char*)d_ws;
  auto carve = [&](size_t bytes)->char*{
    char* p = ws; ws += (bytes + 255) & ~(size_t)255; return p;
  };
  uint16_t* xg     = (uint16_t*)carve((size_t)4 * TBI * II * 2);     // 256 MB
  float*    gxall  = (float*)   carve((size_t)TT * 4 * BB * HH * 4); // 512 MB
  uint16_t* wxT    = (uint16_t*)carve((size_t)4 * HH * II * 2);      // 8 MB
  uint16_t* whT    = (uint16_t*)carve((size_t)4 * HH * HH * 2);      // 8 MB
  float*    bias   = (float*)   carve((size_t)4 * HH * 4);
  uint16_t* hm     = (uint16_t*)carve((size_t)4 * BB * HH * 2);
  float*    cstate = (float*)   carve((size_t)BB * HH * 4);
  (void)ws_size; (void)n_in; (void)in_sizes; (void)out_size;

  // prep
  {
    size_t pairs = (size_t)4 * TBI * II / 2;
    k_prep_x<<<dim3((unsigned)((pairs + 255) / 256)), 256, 0, stream>>>(x, maskx, xg);
  }
  k_prep_wT<<<dim3(4 * 1024 * 1024 / 256), 256, 0, stream>>>(wx, wxT);
  k_prep_wT<<<dim3(4 * 1024 * 1024 / 256), 256, 0, stream>>>(wh, whT);
  k_prep_bias<<<dim3((4 * HH + 255) / 256), 256, 0, stream>>>(bx, bh, bias);
  k_init_state<<<dim3((4 * BB * HH + 255) / 256), 256, 0, stream>>>(h0, c0, maskh, hm, cstate);

  // phase 1: all input GEMMs at once (TDM + LDS double-buffered WMMA GEMM)
  k_gemm_x<<<dim3(TBI / 128, HH / 64, 4), 256, 0, stream>>>(xg, wxT, bias, gxall);

  // phase 2: sequential recurrence
  float* hf = out + (size_t)TT * BB * HH;
  float* cf = hf + (size_t)BB * HH;
  for (int t = 0; t < TT; ++t){
    float* gxt = gxall + (size_t)t * 4 * BB * HH;
    k_gemm_h<<<dim3(1, HH / 16, 4), 128, 0, stream>>>(hm, whT, gxt);
    k_act<<<dim3(BB * HH / 256), 256, 0, stream>>>(
        gxt, maskh, cstate, hm, out + (size_t)t * BB * HH, hf, cf, t == TT - 1 ? 1 : 0);
  }
}
